// VideoMambaBlock_77721728188887
// MI455X (gfx1250) — matrix operand order repaired and checked by reading, not compiled, not run on hardware
//
#include <hip/hip_runtime.h>
#include <hip/hip_bf16.h>

typedef __attribute__((ext_vector_type(16))) __bf16 v16bf;
typedef __attribute__((ext_vector_type(8)))  float  v8f;
typedef __attribute__((ext_vector_type(4)))  float  vf4;

#define D_MODEL 256
#define D_STATE 16
#define D_CONV  4
#define D_INNER 512
#define DT_RANK 16
#define BATCH   2
#define SEQ     4096
#define MTOT    (BATCH * SEQ)            // 8192 rows (b*L + l)
#define NDBL    (DT_RANK + 2 * D_STATE)  // 48

#if defined(__has_builtin)
#if __has_builtin(__builtin_amdgcn_global_load_async_to_lds_b32) && \
    __has_builtin(__builtin_amdgcn_s_wait_asynccnt)
#define USE_ASYNC_LDS 1
#endif
#endif

__device__ __forceinline__ float sigmoidf_(float x) { return 1.f / (1.f + __expf(-x)); }
__device__ __forceinline__ float siluf_(float x)    { return x * sigmoidf_(x); }
__device__ __forceinline__ float softplusf_(float x){ return (x > 20.f) ? x : log1pf(__expf(x)); }

__device__ __forceinline__ void cvt4(v16bf& dst, int base, vf4 v) {
    dst[base + 0] = (__bf16)v.x; dst[base + 1] = (__bf16)v.y;
    dst[base + 2] = (__bf16)v.z; dst[base + 3] = (__bf16)v.w;
}

// One 32-bit element per lane, global -> LDS (async on CDNA5, sync fallback)
__device__ __forceinline__ void copy32_to_lds(float* ldst, const float* gsrc) {
#ifdef USE_ASYNC_LDS
    __builtin_amdgcn_global_load_async_to_lds_b32(
        (__attribute__((address_space(1))) int*)gsrc,
        (__attribute__((address_space(3))) int*)ldst, 0, 0);
#else
    *ldst = *gsrc;
#endif
}
__device__ __forceinline__ void wait_async_lds() {
#ifdef USE_ASYNC_LDS
    __builtin_amdgcn_s_wait_asynccnt(0);
#endif
}

// ---------------------------------------------------------------------------
// GEMM 1: xz[m][n] = sum_k xs[m][k] * W_in[n][k]
//   xs[m][k] = x[b][k][l]  (m = b*SEQ + l), A read strided by SEQ.
//   M=8192, N=1024, K=256.  One wave -> 16x64 C strip (4 WMMA tiles, A reuse).
// ---------------------------------------------------------------------------
__global__ void k_proj_in(const float* __restrict__ X,
                          const float* __restrict__ Win,
                          float* __restrict__ xz) {
    const int NN = 2 * D_INNER;
    int wave = (blockIdx.x * blockDim.x + threadIdx.x) >> 5;
    int lane = threadIdx.x & 31;
    const int tilesM = MTOT / 16;            // 512
    const int strips = NN / 64;              // 16
    int tm = wave % tilesM, ts = wave / tilesM;
    if (ts >= strips) return;                // wave-uniform: EXEC stays full
    int mrow = tm * 16 + (lane & 15);
    int b = mrow >> 12, l = mrow & (SEQ - 1);
    const float* Abase = X + (size_t)b * D_MODEL * SEQ + l;   // + k*SEQ
    const float* Brow[4];
#pragma unroll
    for (int j = 0; j < 4; ++j)
        Brow[j] = Win + (size_t)(ts * 64 + j * 16 + (lane & 15)) * D_MODEL;
    int khA = (lane >> 4) * 16;
    int kbB = (lane >> 4) * 16;
    v8f acc0 = {}, acc1 = {}, acc2 = {}, acc3 = {};
    for (int kk = 0; kk < D_MODEL; kk += 32) {
        v16bf a;
#pragma unroll
        for (int e = 0; e < 8; ++e) {
            a[e]     = (__bf16)Abase[(size_t)(kk + khA + e) * SEQ];
            a[8 + e] = (__bf16)Abase[(size_t)(kk + 8 + khA + e) * SEQ];
        }
        v16bf bm0, bm1, bm2, bm3;
        {
            const vf4* p;
            p = (const vf4*)(Brow[0] + kk + kbB);
            cvt4(bm0, 0, p[0]); cvt4(bm0, 4, p[1]); cvt4(bm0, 8, p[2]); cvt4(bm0, 12, p[3]);
            p = (const vf4*)(Brow[1] + kk + kbB);
            cvt4(bm1, 0, p[0]); cvt4(bm1, 4, p[1]); cvt4(bm1, 8, p[2]); cvt4(bm1, 12, p[3]);
            p = (const vf4*)(Brow[2] + kk + kbB);
            cvt4(bm2, 0, p[0]); cvt4(bm2, 4, p[1]); cvt4(bm2, 8, p[2]); cvt4(bm2, 12, p[3]);
            p = (const vf4*)(Brow[3] + kk + kbB);
            cvt4(bm3, 0, p[0]); cvt4(bm3, 4, p[1]); cvt4(bm3, 8, p[2]); cvt4(bm3, 12, p[3]);
        }
        acc0 = __builtin_amdgcn_wmma_f32_16x16x32_bf16(false, a, false, bm0, (short)0, acc0, false, false);
        acc1 = __builtin_amdgcn_wmma_f32_16x16x32_bf16(false, a, false, bm1, (short)0, acc1, false, false);
        acc2 = __builtin_amdgcn_wmma_f32_16x16x32_bf16(false, a, false, bm2, (short)0, acc2, false, false);
        acc3 = __builtin_amdgcn_wmma_f32_16x16x32_bf16(false, a, false, bm3, (short)0, acc3, false, false);
    }
#pragma unroll
    for (int r = 0; r < 8; ++r) {
        int mr = tm * 16 + r + (lane >> 4) * 8;
        float* obase = xz + (size_t)mr * NN + ts * 64 + (lane & 15);
        obase[0]  = acc0[r];
        obase[16] = acc1[r];
        obase[32] = acc2[r];
        obase[48] = acc3[r];
    }
}

// ---------------------------------------------------------------------------
// Causal depthwise conv (k=4) over L + bias + SiLU:  u[m][d]
// ---------------------------------------------------------------------------
__global__ void k_conv(const float* __restrict__ xz,
                       const float* __restrict__ cw,
                       const float* __restrict__ cb,
                       float* __restrict__ U) {
    int idx = blockIdx.x * blockDim.x + threadIdx.x;
    if (idx >= MTOT * D_INNER) return;
    int d = idx & (D_INNER - 1);
    int m = idx >> 9;
    int b = m >> 12, l = m & (SEQ - 1);
    float s = cb[d];
#pragma unroll
    for (int k = 0; k < D_CONV; ++k) {
        int lp = l + k - (D_CONV - 1);
        if (lp >= 0)
            s += xz[(size_t)(b * SEQ + lp) * (2 * D_INNER) + d] * cw[d * D_CONV + k];
    }
    U[idx] = siluf_(s);
}

// ---------------------------------------------------------------------------
// GEMM 2: x_dbl[m][n] = sum_k u[m][k] * W_x[n][k]   M=8192, N=48, K=512
// One wave covers the whole 16x48 strip (3 WMMA tiles, A reuse).
// ---------------------------------------------------------------------------
__global__ void k_xdbl(const float* __restrict__ U,
                       const float* __restrict__ Wx,
                       float* __restrict__ xdbl) {
    int wave = (blockIdx.x * blockDim.x + threadIdx.x) >> 5;
    int lane = threadIdx.x & 31;
    const int tilesM = MTOT / 16;            // 512
    if (wave >= tilesM) return;
    const float* Arow = U + (size_t)(wave * 16 + (lane & 15)) * D_INNER;
    const float* Brow[3];
#pragma unroll
    for (int j = 0; j < 3; ++j)
        Brow[j] = Wx + (size_t)(j * 16 + (lane & 15)) * D_INNER;
    int khA = (lane >> 4) * 16;
    int kbB = (lane >> 4) * 16;
    v8f acc0 = {}, acc1 = {}, acc2 = {};
    for (int kk = 0; kk < D_INNER; kk += 32) {
        v16bf a;
        {
            const vf4* p = (const vf4*)(Arow + kk + khA);
            cvt4(a, 0, p[0]); cvt4(a, 4, p[1]);
            p = (const vf4*)(Arow + kk + 8 + khA);
            cvt4(a, 8, p[0]); cvt4(a, 12, p[1]);
        }
        v16bf bm0, bm1, bm2;
        {
            const vf4* p;
            p = (const vf4*)(Brow[0] + kk + kbB);
            cvt4(bm0, 0, p[0]); cvt4(bm0, 4, p[1]); cvt4(bm0, 8, p[2]); cvt4(bm0, 12, p[3]);
            p = (const vf4*)(Brow[1] + kk + kbB);
            cvt4(bm1, 0, p[0]); cvt4(bm1, 4, p[1]); cvt4(bm1, 8, p[2]); cvt4(bm1, 12, p[3]);
            p = (const vf4*)(Brow[2] + kk + kbB);
            cvt4(bm2, 0, p[0]); cvt4(bm2, 4, p[1]); cvt4(bm2, 8, p[2]); cvt4(bm2, 12, p[3]);
        }
        acc0 = __builtin_amdgcn_wmma_f32_16x16x32_bf16(false, a, false, bm0, (short)0, acc0, false, false);
        acc1 = __builtin_amdgcn_wmma_f32_16x16x32_bf16(false, a, false, bm1, (short)0, acc1, false, false);
        acc2 = __builtin_amdgcn_wmma_f32_16x16x32_bf16(false, a, false, bm2, (short)0, acc2, false, false);
    }
#pragma unroll
    for (int r = 0; r < 8; ++r) {
        int mr = wave * 16 + r + (lane >> 4) * 8;
        float* obase = xdbl + (size_t)mr * NDBL + (lane & 15);
        obase[0]  = acc0[r];
        obase[16] = acc1[r];
        obase[32] = acc2[r];
    }
}

// ---------------------------------------------------------------------------
// dt expansion + softplus (K=16 -> VALU dot)
// ---------------------------------------------------------------------------
__global__ void k_dt(const float* __restrict__ xdbl,
                     const float* __restrict__ Wdt,
                     const float* __restrict__ bdt,
                     float* __restrict__ dtD) {
    int idx = blockIdx.x * blockDim.x + threadIdx.x;
    if (idx >= MTOT * D_INNER) return;
    int d = idx & (D_INNER - 1);
    int m = idx >> 9;
    float s = bdt[d];
#pragma unroll
    for (int r = 0; r < DT_RANK; ++r)
        s += xdbl[(size_t)m * NDBL + r] * Wdt[d * DT_RANK + r];
    dtD[idx] = softplusf_(s);
}

// ---------------------------------------------------------------------------
// Selective scan: 1024 independent (b,d) recurrences, 8 blocks x 128 channels.
// Double-buffered async global->LDS staging of B/C vectors (next step fetched
// while current step computes); prefetch of next dt/u rows. Fuses +u*Dp and
// the silu(z) gate.
// ---------------------------------------------------------------------------
__global__ void k_scan(const float* __restrict__ xz,
                       const float* __restrict__ U,
                       const float* __restrict__ xdbl,
                       const float* __restrict__ dtD,
                       const float* __restrict__ A_log,
                       const float* __restrict__ Dp,
                       float* __restrict__ Y) {
    const int DCH = 128;
    int b     = blockIdx.x / (D_INNER / DCH);
    int chunk = blockIdx.x % (D_INNER / DCH);
    int d     = chunk * DCH + threadIdx.x;

    float An[D_STATE], h[D_STATE];
#pragma unroll
    for (int n = 0; n < D_STATE; ++n) {
        An[n] = -__expf(A_log[d * D_STATE + n]);
        h[n]  = 0.f;
    }
    float dpd = Dp[d];

    __shared__ float sBC[2][32];   // [buf][ B:0..15 | C:16..31 ]
    int m0 = b * SEQ;
    if (threadIdx.x < 32)
        copy32_to_lds(&sBC[0][threadIdx.x],
                      xdbl + (size_t)m0 * NDBL + DT_RANK + threadIdx.x);
    wait_async_lds();
    __syncthreads();

    for (int l = 0; l < SEQ; ++l) {
        int m = m0 + l;
        int buf = l & 1;
        if (threadIdx.x < 32 && (l + 1) < SEQ)
            copy32_to_lds(&sBC[buf ^ 1][threadIdx.x],
                          xdbl + (size_t)(m + 1) * NDBL + DT_RANK + threadIdx.x);
        if ((l + 1) < SEQ) {
            __builtin_prefetch(dtD + (size_t)(m + 1) * D_INNER + d, 0, 3);
            __builtin_prefetch(U   + (size_t)(m + 1) * D_INNER + d, 0, 3);
        }
        float dt = dtD[(size_t)m * D_INNER + d];
        float uu = U[(size_t)m * D_INNER + d];
        const float* sB = sBC[buf];
        const float* sC = sBC[buf] + D_STATE;
        float yv = 0.f;
#pragma unroll
        for (int n = 0; n < D_STATE; ++n) {
            float dA = __expf(dt * An[n]);
            h[n] = dA * h[n] + dt * sB[n] * uu;
            yv  += h[n] * sC[n];
        }
        yv += uu * dpd;
        float zz = xz[(size_t)m * (2 * D_INNER) + D_INNER + d];
        yv *= zz * sigmoidf_(zz);
        Y[(size_t)m * D_INNER + d] = yv;
        wait_async_lds();
        __syncthreads();
    }
}

// ---------------------------------------------------------------------------
// GEMM 3: out[b][n][l] = sum_k y[m][k] * W_out[n][k]   M=8192, N=256, K=512
// 16x64 strip per wave; stores directly in (B, C, H, W) layout.
// ---------------------------------------------------------------------------
__global__ void k_out(const float* __restrict__ Y,
                      const float* __restrict__ Wout,
                      float* __restrict__ out) {
    int wave = (blockIdx.x * blockDim.x + threadIdx.x) >> 5;
    int lane = threadIdx.x & 31;
    const int tilesM = MTOT / 16;            // 512
    const int strips = D_MODEL / 64;         // 4
    int tm = wave % tilesM, ts = wave / tilesM;
    if (ts >= strips) return;
    const float* Arow = Y + (size_t)(tm * 16 + (lane & 15)) * D_INNER;
    const float* Brow[4];
#pragma unroll
    for (int j = 0; j < 4; ++j)
        Brow[j] = Wout + (size_t)(ts * 64 + j * 16 + (lane & 15)) * D_INNER;
    int khA = (lane >> 4) * 16;
    int kbB = (lane >> 4) * 16;
    v8f acc0 = {}, acc1 = {}, acc2 = {}, acc3 = {};
    for (int kk = 0; kk < D_INNER; kk += 32) {
        v16bf a;
        {
            const vf4* p = (const vf4*)(Arow + kk + khA);
            cvt4(a, 0, p[0]); cvt4(a, 4, p[1]);
            p = (const vf4*)(Arow + kk + 8 + khA);
            cvt4(a, 8, p[0]); cvt4(a, 12, p[1]);
        }
        v16bf bm0, bm1, bm2, bm3;
        {
            const vf4* p;
            p = (const vf4*)(Brow[0] + kk + kbB);
            cvt4(bm0, 0, p[0]); cvt4(bm0, 4, p[1]); cvt4(bm0, 8, p[2]); cvt4(bm0, 12, p[3]);
            p = (const vf4*)(Brow[1] + kk + kbB);
            cvt4(bm1, 0, p[0]); cvt4(bm1, 4, p[1]); cvt4(bm1, 8, p[2]); cvt4(bm1, 12, p[3]);
            p = (const vf4*)(Brow[2] + kk + kbB);
            cvt4(bm2, 0, p[0]); cvt4(bm2, 4, p[1]); cvt4(bm2, 8, p[2]); cvt4(bm2, 12, p[3]);
            p = (const vf4*)(Brow[3] + kk + kbB);
            cvt4(bm3, 0, p[0]); cvt4(bm3, 4, p[1]); cvt4(bm3, 8, p[2]); cvt4(bm3, 12, p[3]);
        }
        acc0 = __builtin_amdgcn_wmma_f32_16x16x32_bf16(false, a, false, bm0, (short)0, acc0, false, false);
        acc1 = __builtin_amdgcn_wmma_f32_16x16x32_bf16(false, a, false, bm1, (short)0, acc1, false, false);
        acc2 = __builtin_amdgcn_wmma_f32_16x16x32_bf16(false, a, false, bm2, (short)0, acc2, false, false);
        acc3 = __builtin_amdgcn_wmma_f32_16x16x32_bf16(false, a, false, bm3, (short)0, acc3, false, false);
    }
#pragma unroll
    for (int r = 0; r < 8; ++r) {
        int mr = tm * 16 + r + (lane >> 4) * 8;
        int bb = mr >> 12, ll = mr & (SEQ - 1);
        float* obase = out + (size_t)bb * D_MODEL * SEQ + ll;
        int nc = ts * 64 + (lane & 15);
        obase[(size_t)(nc)      * SEQ] = acc0[r];
        obase[(size_t)(nc + 16) * SEQ] = acc1[r];
        obase[(size_t)(nc + 32) * SEQ] = acc2[r];
        obase[(size_t)(nc + 48) * SEQ] = acc3[r];
    }
}

extern "C" void kernel_launch(void* const* d_in, const int* in_sizes, int n_in,
                              void* d_out, int out_size, void* d_ws, size_t ws_size,
                              hipStream_t stream) {
    (void)in_sizes; (void)n_in; (void)out_size; (void)ws_size;
    const float* X      = (const float*)d_in[0];
    const float* W_in   = (const float*)d_in[1];
    const float* conv_w = (const float*)d_in[2];
    const float* conv_b = (const float*)d_in[3];
    const float* W_x    = (const float*)d_in[4];
    const float* W_dt   = (const float*)d_in[5];
    const float* b_dt   = (const float*)d_in[6];
    const float* A_log  = (const float*)d_in[7];
    const float* Dp     = (const float*)d_in[8];
    const float* W_out  = (const float*)d_in[9];
    float* out = (float*)d_out;

    float* xz   = (float*)d_ws;                          // 8192*1024 f (32 MB)
    float* U    = xz   + (size_t)MTOT * 2 * D_INNER;     // 8192*512  f (16 MB)
    float* xdbl = U    + (size_t)MTOT * D_INNER;         // 8192*48   f (1.5 MB)
    float* dtD  = xdbl + (size_t)MTOT * NDBL;            // 8192*512  f (16 MB)
    float* Y    = dtD  + (size_t)MTOT * D_INNER;         // 8192*512  f (16 MB)

    {   // in-projection: 512 M-tiles x 16 strips = 8192 waves
        int waves = (MTOT / 16) * ((2 * D_INNER) / 64);
        k_proj_in<<<waves * 32 / 256, 256, 0, stream>>>(X, W_in, xz);
    }
    {
        int n = MTOT * D_INNER;
        k_conv<<<(n + 255) / 256, 256, 0, stream>>>(xz, conv_w, conv_b, U);
    }
    {   // x_dbl: 512 waves (one 16x48 strip each)
        int waves = MTOT / 16;
        k_xdbl<<<waves * 32 / 256, 256, 0, stream>>>(U, W_x, xdbl);
    }
    {
        int n = MTOT * D_INNER;
        k_dt<<<(n + 255) / 256, 256, 0, stream>>>(xdbl, W_dt, b_dt, dtD);
    }
    k_scan<<<BATCH * (D_INNER / 128), 128, 0, stream>>>(xz, U, xdbl, dtD, A_log, Dp, Y);
    {   // out-projection: 512 M-tiles x 4 strips = 2048 waves
        int waves = (MTOT / 16) * (D_MODEL / 64);
        k_out<<<waves * 32 / 256, 256, 0, stream>>>(Y, W_out, out);
    }
}
